// CustomMultiHeadAttention_21732534518437
// MI455X (gfx1250) — compile-verified, matrix operand order checked by reading, not run on hardware
//
#include <hip/hip_runtime.h>
#include <hip/hip_bf16.h>

// ---------------------------------------------------------------------------
// Causal multi-head attention forward for MI455X (gfx1250, wave32, WMMA).
//   B=2, T=2048, C=1024, H=16, Dh=64.
// Pipeline:
//   [cvt f32->f16 pre-pass for X and all weights]  (bandwidth bound, ~3us)
//   [qkv_gemm: pure f16 WMMA, 32x64 per wave]
//   [flash attention: WMMA f16, online softmax, wave-private LDS P-staging]
//   [out_gemm: pure f16 WMMA, f32 epilogue]
// ---------------------------------------------------------------------------

#define EMBED 1024
#define SEQ_T 2048
#define BATCH 2
#define HEADS 16
#define DHEAD 64

typedef __attribute__((ext_vector_type(16))) _Float16 v16h;
typedef __attribute__((ext_vector_type(8)))  _Float16 v8h;
typedef __attribute__((ext_vector_type(8)))  float    v8f;
typedef __attribute__((ext_vector_type(4)))  float    v4f;

// ---------------------------------------------------------------------------
// Fragment loader: 16-bit A-matrix 16x32 (MxK) per ISA 7.12.2.
// lane L -> row = L%16; halves cover K = {8g..8g+7, 16+8g..16+8g+7}, g=L/16.
// B fragments use the same loader on B^T rows (column-major B per ISA), which
// for X@W^T / Q@K^T / P@V^T is a plain row-major strided load.
// ---------------------------------------------------------------------------
__device__ inline v16h frag_f16(const _Float16* __restrict__ base, int ld, int lane) {
  const int r = lane & 15;
  const int g = lane >> 4;
  const _Float16* p = base + (size_t)r * ld + 8 * g;
  v8h lo = *(const v8h*)(p);
  v8h hi = *(const v8h*)(p + 16);
  v16h out;
#pragma unroll
  for (int i = 0; i < 8; ++i) {
    out[i]     = lo[i];
    out[8 + i] = hi[i];
  }
  return out;
}

__device__ inline v8f wmma16x16x32(v16h a, v16h b, v8f c) {
  // (neg_a, A, neg_b, B, c_mod, C, reuse_a, reuse_b)
  return __builtin_amdgcn_wmma_f32_16x16x32_f16(false, a, false, b, (short)0, c,
                                                false, false);
}

// ---------------------------------------------------------------------------
// Kernel 0: elementwise f32 -> f16 (vectorized 8/thread).  n % 8 == 0.
// ---------------------------------------------------------------------------
__global__ __launch_bounds__(256) void cvt_f32_to_f16(
    const float* __restrict__ src, _Float16* __restrict__ dst, int n)
{
  const int i = (blockIdx.x * 256 + threadIdx.x) * 8;
  if (i >= n) return;
  v4f a = *(const v4f*)(src + i);
  v4f b = *(const v4f*)(src + i + 4);
  v8h o;
#pragma unroll
  for (int j = 0; j < 4; ++j) {
    o[j]     = (_Float16)a[j];
    o[4 + j] = (_Float16)b[j];
  }
  *(v8h*)(dst + i) = o;
}

// ---------------------------------------------------------------------------
// Kernel 1: fused QKV projection, all-f16 operands.
// Block = 256 thr (8 waves) -> 128(M) x 128(N) tile; wave = 32x64
// (2 M-frags x 4 N-frags = 8 WMMA per 32-deep k-step, 12 b128 loads).
// Q,K stored (B,H,T,Dh) f16;  V stored transposed (B,H,Dh,T) f16.
// ---------------------------------------------------------------------------
__global__ __launch_bounds__(256) void qkv_gemm(
    const _Float16* __restrict__ Xh,
    const _Float16* __restrict__ Wqh, const float* __restrict__ bq,
    const _Float16* __restrict__ Wkh, const float* __restrict__ bk,
    const _Float16* __restrict__ Wvh, const float* __restrict__ bv,
    _Float16* __restrict__ Qh, _Float16* __restrict__ Kh,
    _Float16* __restrict__ Vt)
{
  const int lane = threadIdx.x & 31;
  const int wave = threadIdx.x >> 5;
  const int r = lane & 15;
  const int g = lane >> 4;
  const int which = blockIdx.z;
  const _Float16* __restrict__ W = (which == 0) ? Wqh : (which == 1) ? Wkh : Wvh;
  const float* __restrict__ bias = (which == 0) ? bq : (which == 1) ? bk : bv;

  const int m0 = blockIdx.x * 128 + (wave & 3) * 32;   // row of X (b*T + t)
  const int n0 = blockIdx.y * 128 + (wave >> 2) * 64;  // output feature

  v8f zero = {};
  v8f acc[2][4];
#pragma unroll
  for (int mt = 0; mt < 2; ++mt)
#pragma unroll
    for (int nt = 0; nt < 4; ++nt) acc[mt][nt] = zero;

  const _Float16* __restrict__ Xb = Xh + (size_t)m0 * EMBED;
  for (int k0 = 0; k0 < EMBED; k0 += 32) {
    __builtin_prefetch(Xb + k0 + 64, 0, 1);
    v16h a0 = frag_f16(Xb + k0, EMBED, lane);
    v16h a1 = frag_f16(Xb + (size_t)16 * EMBED + k0, EMBED, lane);
#pragma unroll
    for (int nt = 0; nt < 4; ++nt) {
      v16h bf = frag_f16(W + (size_t)(n0 + nt * 16) * EMBED + k0, EMBED, lane);
      acc[0][nt] = wmma16x16x32(a0, bf, acc[0][nt]);
      acc[1][nt] = wmma16x16x32(a1, bf, acc[1][nt]);
    }
  }

#pragma unroll
  for (int nt = 0; nt < 4; ++nt) {
    const int n = n0 + nt * 16 + r;
    const float bb = bias[n];
    const int h = n >> 6;        // head
    const int d = n & 63;        // dim within head
#pragma unroll
    for (int mt = 0; mt < 2; ++mt) {
#pragma unroll
      for (int v = 0; v < 8; ++v) {
        const int row = m0 + mt * 16 + v + 8 * g;  // C frag: M = vgpr + 8*(lane/16)
        const int b = row >> 11;                   // row / T
        const int t = row & (SEQ_T - 1);
        const float val = acc[mt][nt][v] + bb;
        const size_t bh = (size_t)(b * HEADS + h);
        if (which == 0)      Qh[(bh * SEQ_T + t) * DHEAD + d] = (_Float16)val;
        else if (which == 1) Kh[(bh * SEQ_T + t) * DHEAD + d] = (_Float16)val;
        else                 Vt[(bh * DHEAD + d) * SEQ_T + t] = (_Float16)val;
      }
    }
  }
}

// ---------------------------------------------------------------------------
// Kernel 2: causal flash attention.  Grid: (T/64, B*H); block = 128 thr
// (4 waves).  Each wave owns a 16-query tile and streams 32-key blocks up to
// the diagonal.  QK^T: 4 WMMA / block; PV: 4 WMMA / block.  P (f32 C-frags)
// is staged through per-wave LDS to become an f16 A-fragment.
// ---------------------------------------------------------------------------
__global__ __launch_bounds__(128) void attn_kernel(
    const _Float16* __restrict__ Qh, const _Float16* __restrict__ Kh,
    const _Float16* __restrict__ Vt, _Float16* __restrict__ Ah)
{
  __shared__ _Float16 Pl_all[4][16 * 32];   // 1KB per wave

  const int lane = threadIdx.x & 31;
  const int wave = threadIdx.x >> 5;
  const int r = lane & 15;
  const int g = lane >> 4;
  const int bh = blockIdx.y;                // 0..B*H-1
  const int b = bh >> 4;
  const int h = bh & 15;
  const int qbase = blockIdx.x * 64 + wave * 16;
  _Float16* __restrict__ Pl = Pl_all[wave];

  const float scale = 0.125f;               // 1/sqrt(64)
  const float NEG = -3.0e38f;

  const _Float16* __restrict__ Qb  = Qh + ((size_t)bh * SEQ_T + qbase) * DHEAD;
  const _Float16* __restrict__ Kbh = Kh + (size_t)bh * SEQ_T * DHEAD;
  const _Float16* __restrict__ Vbh = Vt + (size_t)bh * DHEAD * SEQ_T;

  // Q fragments (d = 0..31, 32..63) live in registers for the whole kernel.
  const v16h qa0 = frag_f16(Qb, DHEAD, lane);
  const v16h qa1 = frag_f16(Qb + 32, DHEAD, lane);

  v8f zero = {};
  v8f o[4];
#pragma unroll
  for (int dt = 0; dt < 4; ++dt) o[dt] = zero;

  float mrow[8], lrow[8];
#pragma unroll
  for (int v = 0; v < 8; ++v) { mrow[v] = -3.0e38f; lrow[v] = 0.0f; }

  const int nblocks = (qbase >> 5) + 1;     // 32-key blocks up to diagonal
  for (int j = 0; j < nblocks; ++j) {
    const int kb = j * 32;

    // ---- S = (Q K^T) * scale, 16 queries x 32 keys -----------------------
    v8f s[2];
#pragma unroll
    for (int nt = 0; nt < 2; ++nt) {
      const _Float16* Kt = Kbh + (size_t)(kb + nt * 16) * DHEAD;
      v8f acc = wmma16x16x32(qa0, frag_f16(Kt, DHEAD, lane), zero);
      s[nt] = wmma16x16x32(qa1, frag_f16(Kt + 32, DHEAD, lane), acc);
    }

    // ---- online softmax: per-row (m = v + 8g) reductions over 16 lanes ---
    float corr[8];
#pragma unroll
    for (int v = 0; v < 8; ++v) {
      const int m = v + 8 * g;
      const int qg = qbase + m;
      float s0 = s[0][v] * scale;
      float s1 = s[1][v] * scale;
      if (kb + r > qg)      s0 = NEG;       // causal mask
      if (kb + 16 + r > qg) s1 = NEG;
      float mx = fmaxf(s0, s1);
#pragma unroll
      for (int off = 8; off > 0; off >>= 1) mx = fmaxf(mx, __shfl_xor(mx, off, 32));
      const float nm = fmaxf(mrow[v], mx);
      corr[v] = __expf(mrow[v] - nm);
      mrow[v] = nm;
      const float p0 = __expf(s0 - nm);
      const float p1 = __expf(s1 - nm);
      float rs = p0 + p1;
#pragma unroll
      for (int off = 8; off > 0; off >>= 1) rs += __shfl_xor(rs, off, 32);
      lrow[v] = lrow[v] * corr[v] + rs;
      // stage P row-major 16x32 f16 in LDS (wave-private tile)
      Pl[m * 32 + r]      = (_Float16)p0;
      Pl[m * 32 + 16 + r] = (_Float16)p1;
    }

    // rescale running output by exp(old_max - new_max)
#pragma unroll
    for (int dt = 0; dt < 4; ++dt)
#pragma unroll
      for (int v = 0; v < 8; ++v) o[dt][v] *= corr[v];

    // wave-private LDS tile: only need DS-counter drain, no workgroup barrier
    asm volatile("s_wait_dscnt 0" ::: "memory");
    const v16h pa = frag_f16(Pl, 32, lane);

    // ---- O += P @ V  (V stored transposed: rows = d, cols = t) -----------
#pragma unroll
    for (int dt = 0; dt < 4; ++dt) {
      v16h bf = frag_f16(Vbh + (size_t)(dt * 16) * SEQ_T + kb, SEQ_T, lane);
      o[dt] = wmma16x16x32(pa, bf, o[dt]);
    }
  }

  // ---- normalize and write (B,T,C) f16 ----------------------------------
#pragma unroll
  for (int dt = 0; dt < 4; ++dt) {
#pragma unroll
    for (int v = 0; v < 8; ++v) {
      const int m = v + 8 * g;
      const int t = qbase + m;
      const float val = o[dt][v] / lrow[v];
      Ah[((size_t)b * SEQ_T + t) * EMBED + h * DHEAD + dt * 16 + r] = (_Float16)val;
    }
  }
}

// ---------------------------------------------------------------------------
// Kernel 3: output projection  out = Attn @ Wo^T + bo  (f32 result).
// Same tiling as kernel 1 (32x64 per wave, 128x128 per block), all f16.
// ---------------------------------------------------------------------------
__global__ __launch_bounds__(256) void out_gemm(
    const _Float16* __restrict__ A, const _Float16* __restrict__ Woh,
    const float* __restrict__ bo, float* __restrict__ out)
{
  const int lane = threadIdx.x & 31;
  const int wave = threadIdx.x >> 5;
  const int r = lane & 15;
  const int g = lane >> 4;

  const int m0 = blockIdx.x * 128 + (wave & 3) * 32;
  const int n0 = blockIdx.y * 128 + (wave >> 2) * 64;

  v8f zero = {};
  v8f acc[2][4];
#pragma unroll
  for (int mt = 0; mt < 2; ++mt)
#pragma unroll
    for (int nt = 0; nt < 4; ++nt) acc[mt][nt] = zero;

  const _Float16* __restrict__ Ab = A + (size_t)m0 * EMBED;
  for (int k0 = 0; k0 < EMBED; k0 += 32) {
    __builtin_prefetch(Ab + k0 + 64, 0, 1);
    v16h a0 = frag_f16(Ab + k0, EMBED, lane);
    v16h a1 = frag_f16(Ab + (size_t)16 * EMBED + k0, EMBED, lane);
#pragma unroll
    for (int nt = 0; nt < 4; ++nt) {
      v16h bf = frag_f16(Woh + (size_t)(n0 + nt * 16) * EMBED + k0, EMBED, lane);
      acc[0][nt] = wmma16x16x32(a0, bf, acc[0][nt]);
      acc[1][nt] = wmma16x16x32(a1, bf, acc[1][nt]);
    }
  }

#pragma unroll
  for (int nt = 0; nt < 4; ++nt) {
    const int n = n0 + nt * 16 + r;
    const float bb = bo[n];
#pragma unroll
    for (int mt = 0; mt < 2; ++mt) {
#pragma unroll
      for (int v = 0; v < 8; ++v) {
        const int row = m0 + mt * 16 + v + 8 * g;
        out[(size_t)row * EMBED + n] = acc[mt][nt][v] + bb;
      }
    }
  }
}

// ---------------------------------------------------------------------------
extern "C" void kernel_launch(void* const* d_in, const int* in_sizes, int n_in,
                              void* d_out, int out_size, void* d_ws, size_t ws_size,
                              hipStream_t stream) {
  const float* x  = (const float*)d_in[0];
  const float* Wq = (const float*)d_in[1];
  const float* bq = (const float*)d_in[2];
  const float* Wk = (const float*)d_in[3];
  const float* bk = (const float*)d_in[4];
  const float* Wv = (const float*)d_in[5];
  const float* bv = (const float*)d_in[6];
  const float* Wo = (const float*)d_in[7];
  const float* bo = (const float*)d_in[8];
  float* out = (float*)d_out;

  _Float16* ws = (_Float16*)d_ws;
  const size_t NACT = (size_t)BATCH * SEQ_T * EMBED;   // 4M halves = 8 MB
  const size_t NWGT = (size_t)EMBED * EMBED;           // 1M halves = 2 MB
  _Float16* Xh  = ws;                                  // (B*T, C)  f16
  _Float16* Wqh = Xh + NACT;
  _Float16* Wkh = Wqh + NWGT;
  _Float16* Wvh = Wkh + NWGT;
  _Float16* Woh = Wvh + NWGT;
  _Float16* Qh  = Woh + NWGT;        // (B,H,T,Dh) f16
  _Float16* Kh  = Qh + NACT;         // (B,H,T,Dh) f16
  _Float16* Vt  = Kh + NACT;         // (B,H,Dh,T) f16 (transposed)
  _Float16* Ah  = Vt + NACT;         // (B,T,C)    f16

  // --- f32 -> f16 pre-pass (bandwidth bound) ------------------------------
  cvt_f32_to_f16<<<dim3((int)(NACT / (8 * 256))), 256, 0, stream>>>(x, Xh, (int)NACT);
  cvt_f32_to_f16<<<dim3((int)(NWGT / (8 * 256))), 256, 0, stream>>>(Wq, Wqh, (int)NWGT);
  cvt_f32_to_f16<<<dim3((int)(NWGT / (8 * 256))), 256, 0, stream>>>(Wk, Wkh, (int)NWGT);
  cvt_f32_to_f16<<<dim3((int)(NWGT / (8 * 256))), 256, 0, stream>>>(Wv, Wvh, (int)NWGT);
  cvt_f32_to_f16<<<dim3((int)(NWGT / (8 * 256))), 256, 0, stream>>>(Wo, Woh, (int)NWGT);

  // --- QKV: M=4096 (x/128=32), N=1024 (y/128=8), z selects q/k/v ----------
  qkv_gemm<<<dim3(32, 8, 3), 256, 0, stream>>>(Xh, Wqh, bq, Wkh, bk, Wvh, bv,
                                               Qh, Kh, Vt);
  // --- Attention: (T/64 q-tiles, B*H heads), 4 waves per block ------------
  attn_kernel<<<dim3(SEQ_T / 64, BATCH * HEADS), 128, 0, stream>>>(Qh, Kh, Vt, Ah);
  // --- Output projection: M=4096, N=1024 ----------------------------------
  out_gemm<<<dim3(32, 8), 256, 0, stream>>>(Ah, Woh, bo, out);
}
